// Transformer3DBlock_27968827031825
// MI455X (gfx1250) — matrix operand order repaired and checked by reading, not compiled, not run on hardware
//
#include <hip/hip_runtime.h>
#include <hip/hip_bf16.h>

#define DIM   192
#define HEADS 6
#define HD    32
#define WT    144
#define NW_   64
#define NWIN  960              // (B*NLON)*... = 15 * 64
#define NTOK  (NWIN * WT)      // 138240
#define VPAD  160              // K-padded token dim for P@V
#define MLP   768
#define QSCALE 0.17677669529663687f  // 32^-0.5

typedef __attribute__((ext_vector_type(16))) _Float16 v16h;
typedef __attribute__((ext_vector_type(8)))  _Float16 v8h;
typedef __attribute__((ext_vector_type(8)))  float    v8f;

__device__ __forceinline__ v16h frag_cat(v8h lo, v8h hi) {
  return __builtin_shufflevector(lo, hi, 0,1,2,3,4,5,6,7,8,9,10,11,12,13,14,15);
}

// A fragment (16xK=32, row-major source, leading dim lda in halfs).
// Lane<16: row=lane, K {k0..k0+7, k0+16..k0+23}; lane>=16: +8 on both K runs.
__device__ __forceinline__ v16h load_frag_a(const _Float16* A, int lda, int row0, int k0, int lane) {
  int r  = row0 + (lane & 15);
  int kb = k0 + ((lane & 16) ? 8 : 0);
  const _Float16* p = A + (size_t)r * lda + kb;
  v8h lo = *(const v8h*)p;
  v8h hi = *(const v8h*)(p + 16);
  return frag_cat(lo, hi);
}

// B fragment for B(k,n) = W[n][k], W row-major [N][ldw].
// Lane holds column n=n0+(lane&15), K {k0..k0+7, k0+16..k0+23}; lane>=16: +8 on both K runs.
__device__ __forceinline__ v16h load_frag_b(const _Float16* W, int ldw, int n0, int k0, int lane) {
  int n  = n0 + (lane & 15);
  int kb = k0 + ((lane & 16) ? 8 : 0);
  const _Float16* p = W + (size_t)n * ldw + kb;
  v8h lo = *(const v8h*)p;
  v8h hi = *(const v8h*)(p + 16);
  return frag_cat(lo, hi);
}

__device__ __forceinline__ v8f wmma_f16(v16h a, v16h b, v8f c) {
  return __builtin_amdgcn_wmma_f32_16x16x32_f16(false, a, false, b, (short)0, c, false, false);
}

// ---------------- weight convert ----------------
__global__ void f32_to_f16(const float* __restrict__ s, _Float16* __restrict__ d, int n) {
  int i = blockIdx.x * 256 + threadIdx.x;
  if (i < n) d[i] = (_Float16)s[i];
}

// ---------------- LN1 + roll(-1,-3,-6) + window partition ----------------
__global__ void __launch_bounds__(128) ln1_shift_kernel(const float* __restrict__ x,
    const float* __restrict__ g, const float* __restrict__ b, _Float16* __restrict__ outw) {
  int wave = threadIdx.x >> 5, lane = threadIdx.x & 31;
  int gid = blockIdx.x * 4 + wave;                // window-order token id
  int win = gid / WT, t = gid % WT;
  int lb = win / NW_, nw = win % NW_, pb = nw >> 4, latb = nw & 15;
  int zi = t / 72, hi = (t / 12) % 6, wi = t % 12;
  int p = pb * 2 + zi, la = latb * 6 + hi, lo = lb * 12 + wi;
  int sp = (p + 1) & 7;
  int sla = la + 3; if (sla >= 96) sla -= 96;
  int slo = lo + 6; if (slo >= 180) slo -= 180;
  size_t src = ((size_t)(sp * 96 + sla) * 180 + slo) * DIM;
  int c0 = lane * 6;
  float v[6], s = 0.f, sq = 0.f;
  #pragma unroll
  for (int i = 0; i < 6; ++i) { v[i] = x[src + c0 + i]; s += v[i]; sq += v[i] * v[i]; }
  #pragma unroll
  for (int off = 16; off; off >>= 1) { s += __shfl_xor(s, off); sq += __shfl_xor(sq, off); }
  float mean = s * (1.f / DIM);
  float var  = sq * (1.f / DIM) - mean * mean;
  float rstd = rsqrtf(var + 1e-5f);
  size_t dst = (size_t)gid * DIM + c0;
  #pragma unroll
  for (int i = 0; i < 6; ++i)
    outw[dst + i] = (_Float16)((v[i] - mean) * rstd * g[c0 + i] + b[c0 + i]);
}

// ---------------- LN2 (plain order) ----------------
__global__ void __launch_bounds__(128) ln2_kernel(const float* __restrict__ x,
    const float* __restrict__ g, const float* __restrict__ b, _Float16* __restrict__ y) {
  int wave = threadIdx.x >> 5, lane = threadIdx.x & 31;
  int gid = blockIdx.x * 4 + wave;
  size_t src = (size_t)gid * DIM;
  int c0 = lane * 6;
  float v[6], s = 0.f, sq = 0.f;
  #pragma unroll
  for (int i = 0; i < 6; ++i) { v[i] = x[src + c0 + i]; s += v[i]; sq += v[i] * v[i]; }
  #pragma unroll
  for (int off = 16; off; off >>= 1) { s += __shfl_xor(s, off); sq += __shfl_xor(sq, off); }
  float mean = s * (1.f / DIM);
  float var  = sq * (1.f / DIM) - mean * mean;
  float rstd = rsqrtf(var + 1e-5f);
  #pragma unroll
  for (int i = 0; i < 6; ++i)
    y[src + c0 + i] = (_Float16)((v[i] - mean) * rstd * g[c0 + i] + b[c0 + i]);
}

// ---------------- QKV GEMM: [NTOK,192] x [192,576] ----------------
__global__ void __launch_bounds__(256) qkv_gemm(const _Float16* __restrict__ Aw,
    const _Float16* __restrict__ Wq, const float* __restrict__ qb,
    _Float16* __restrict__ Q, _Float16* __restrict__ K, _Float16* __restrict__ V) {
  int lane = threadIdx.x & 31, wave = threadIdx.x >> 5;
  int row0 = blockIdx.x * 128 + wave * 16;
  int col0 = blockIdx.y * 64;
  v8f acc[4] = {};
  for (int ks = 0; ks < 6; ++ks) {
    v16h a = load_frag_a(Aw, DIM, row0, ks * 32, lane);
    #pragma unroll
    for (int j = 0; j < 4; ++j) {
      v16h b = load_frag_b(Wq, DIM, col0 + j * 16, ks * 32, lane);
      acc[j] = wmma_f16(a, b, acc[j]);
    }
  }
  int rbase = row0 + ((lane & 16) ? 8 : 0);
  #pragma unroll
  for (int j = 0; j < 4; ++j) {
    int f = col0 + j * 16 + (lane & 15);
    int which = f / DIM;
    int head  = (f % DIM) / HD;
    int d     = f % HD;
    float bias = qb[f];
    #pragma unroll
    for (int r = 0; r < 8; ++r) {
      int tok = rbase + r;
      int win = tok / WT, t = tok % WT;
      float val = acc[j][r] + bias;
      if (which == 0)
        Q[(((size_t)win * HEADS + head) * WT + t) * HD + d] = (_Float16)(val * QSCALE);
      else if (which == 1)
        K[(((size_t)win * HEADS + head) * WT + t) * HD + d] = (_Float16)val;
      else
        V[(((size_t)win * HEADS + head) * HD + d) * VPAD + t] = (_Float16)val;
    }
  }
}

// ---------------- attention ----------------
#define SPITCH 148
#define SBYTES (WT * SPITCH * 4)
#define PPITCH 160
#define SMEM_ATTN (SBYTES + WT * PPITCH * 2)

__device__ __forceinline__ int region_code(int t, int lb, int pb, int latb) {
  int zi = t / 72, hi = (t / 12) % 6, wi = t % 12;
  int p = pb * 2 + zi, la = latb * 6 + hi, lo = lb * 12 + wi;
  int pr  = (p  <   6) ? 0 : ((p  ==  6) ? 1 : 2);
  int lar = (la <  90) ? 0 : ((la <  93) ? 1 : 2);
  int lor = (lo < 174) ? 0 : 1;
  return pr * 9 + lar * 3 + lor;
}
__device__ __forceinline__ int pos_idx(int i, int j) {
  int zi = i / 72, hi = (i / 12) % 6, w  = i % 12;
  int zj = j / 72, hj = (j / 12) % 6, w2 = j % 12;
  return (zi + 2 * zj) * 828 + (hi + 6 * hj) * 23 + (w - w2 + 11);
}

__global__ void __launch_bounds__(288) attn_kernel(const _Float16* __restrict__ Q,
    const _Float16* __restrict__ K, const _Float16* __restrict__ Vt,
    const float* __restrict__ btab, _Float16* __restrict__ O) {
  extern __shared__ char smem[];
  float*    S = (float*)smem;                       // [144][148] fp32 scores
  _Float16* P = (_Float16*)(smem + SBYTES);         // [144][160] f16 probs (K-padded)
  int wh = blockIdx.x;
  int win = wh / HEADS, head = wh % HEADS;
  int lb = win / NW_, nw = win % NW_, pb = nw >> 4, latb = nw & 15;
  const _Float16* Qw = Q  + (size_t)wh * WT * HD;
  const _Float16* Kw = K  + (size_t)wh * WT * HD;
  const _Float16* Vw = Vt + (size_t)wh * HD * VPAD;
  int wave = threadIdx.x >> 5, lane = threadIdx.x & 31;
  int row0 = wave * 16;
  int rbase = row0 + ((lane & 16) ? 8 : 0);

  // S = (Q*scale) K^T + bias + mask  (hd=32 -> single K-step)
  v16h a = load_frag_a(Qw, HD, row0, 0, lane);
  int regi[8];
  #pragma unroll
  for (int r = 0; r < 8; ++r) regi[r] = region_code(rbase + r, lb, pb, latb);
  for (int nt = 0; nt < 9; ++nt) {
    v16h b = load_frag_b(Kw, HD, nt * 16, 0, lane);
    v8f c = {};
    c = wmma_f16(a, b, c);
    int col  = nt * 16 + (lane & 15);
    int regj = region_code(col, lb, pb, latb);
    #pragma unroll
    for (int r = 0; r < 8; ++r) {
      int row = rbase + r;
      int idx = pos_idx(row, col);
      float bias = btab[((size_t)idx * NW_ + nw) * HEADS + head];
      float m = (regi[r] == regj) ? 0.0f : -100.0f;
      S[row * SPITCH + col] = c[r] + bias + m;
    }
  }
  // per-row softmax; each wave owns exactly its 16 rows -> no barriers needed
  for (int r = 0; r < 16; ++r) {
    int row = row0 + r;
    float mx = -1e30f;
    for (int cidx = lane; cidx < WT; cidx += 32) mx = fmaxf(mx, S[row * SPITCH + cidx]);
    #pragma unroll
    for (int off = 16; off; off >>= 1) mx = fmaxf(mx, __shfl_xor(mx, off));
    float sum = 0.f;
    for (int cidx = lane; cidx < WT; cidx += 32) {
      float e = __expf(S[row * SPITCH + cidx] - mx);
      S[row * SPITCH + cidx] = e;
      sum += e;
    }
    #pragma unroll
    for (int off = 16; off; off >>= 1) sum += __shfl_xor(sum, off);
    float inv = 1.0f / sum;
    for (int cidx = lane; cidx < WT; cidx += 32)
      P[row * PPITCH + cidx] = (_Float16)(S[row * SPITCH + cidx] * inv);
    if (lane < 16) P[row * PPITCH + WT + lane] = (_Float16)0.f;  // zero K-pad
  }
  // O = P @ V  (K padded to 160 -> 5 K-steps, N=32 -> 2 tiles)
  #pragma unroll
  for (int nt = 0; nt < 2; ++nt) {
    v8f c = {};
    for (int ks = 0; ks < 5; ++ks) {
      v16h pa = load_frag_a(P, PPITCH, row0, ks * 32, lane);
      v16h vb = load_frag_b(Vw, VPAD, nt * 16, ks * 32, lane);
      c = wmma_f16(pa, vb, c);
    }
    int d = nt * 16 + (lane & 15);
    #pragma unroll
    for (int r = 0; r < 8; ++r)
      O[((size_t)win * WT + rbase + r) * DIM + head * HD + d] = (_Float16)c[r];
  }
}

// ---------------- proj GEMM + window reverse + roll(+1,+3,+6) + residual ----------------
__global__ void __launch_bounds__(256) proj_gemm(const _Float16* __restrict__ O,
    const _Float16* __restrict__ Wp, const float* __restrict__ pbias,
    const float* __restrict__ x, float* __restrict__ out) {
  int lane = threadIdx.x & 31, wave = threadIdx.x >> 5;
  int row0 = blockIdx.x * 128 + wave * 16;
  int col0 = blockIdx.y * 64;
  v8f acc[4] = {};
  for (int ks = 0; ks < 6; ++ks) {
    v16h a = load_frag_a(O, DIM, row0, ks * 32, lane);
    #pragma unroll
    for (int j = 0; j < 4; ++j) {
      v16h b = load_frag_b(Wp, DIM, col0 + j * 16, ks * 32, lane);
      acc[j] = wmma_f16(a, b, acc[j]);
    }
  }
  int rbase = row0 + ((lane & 16) ? 8 : 0);
  #pragma unroll
  for (int j = 0; j < 4; ++j) {
    int f = col0 + j * 16 + (lane & 15);
    float bv = pbias[f];
    #pragma unroll
    for (int r = 0; r < 8; ++r) {
      int tok = rbase + r;
      int win = tok / WT, t = tok % WT;
      int lb = win / NW_, nw = win % NW_, pb = nw >> 4, latb = nw & 15;
      int zi = t / 72, hi = (t / 12) % 6, wi = t % 12;
      int p = pb * 2 + zi, la = latb * 6 + hi, lo = lb * 12 + wi;
      int op  = (p + 1) & 7;
      int ola = la + 3; if (ola >= 96)  ola -= 96;
      int olo = lo + 6; if (olo >= 180) olo -= 180;
      size_t dst = ((size_t)(op * 96 + ola) * 180 + olo) * DIM + f;
      out[dst] = x[dst] + acc[j][r] + bv;
    }
  }
}

// ---------------- FC1 + exact GELU ----------------
__global__ void __launch_bounds__(256) fc1_gemm(const _Float16* __restrict__ Y,
    const _Float16* __restrict__ W1, const float* __restrict__ b1, _Float16* __restrict__ H) {
  int lane = threadIdx.x & 31, wave = threadIdx.x >> 5;
  int row0 = blockIdx.x * 128 + wave * 16;
  int col0 = blockIdx.y * 64;
  v8f acc[4] = {};
  for (int ks = 0; ks < 6; ++ks) {
    v16h a = load_frag_a(Y, DIM, row0, ks * 32, lane);
    #pragma unroll
    for (int j = 0; j < 4; ++j) {
      v16h b = load_frag_b(W1, DIM, col0 + j * 16, ks * 32, lane);
      acc[j] = wmma_f16(a, b, acc[j]);
    }
  }
  int rbase = row0 + ((lane & 16) ? 8 : 0);
  #pragma unroll
  for (int j = 0; j < 4; ++j) {
    int f = col0 + j * 16 + (lane & 15);
    float bv = b1[f];
    #pragma unroll
    for (int r = 0; r < 8; ++r) {
      float u = acc[j][r] + bv;
      float gl = 0.5f * u * (1.0f + erff(u * 0.70710678118654752f));
      H[(size_t)(rbase + r) * MLP + f] = (_Float16)gl;
    }
  }
}

// ---------------- FC2 + residual into d_out ----------------
__global__ void __launch_bounds__(256) fc2_gemm(const _Float16* __restrict__ H,
    const _Float16* __restrict__ W2, const float* __restrict__ b2, float* __restrict__ out) {
  int lane = threadIdx.x & 31, wave = threadIdx.x >> 5;
  int row0 = blockIdx.x * 128 + wave * 16;
  int col0 = blockIdx.y * 64;
  v8f acc[4] = {};
  for (int ks = 0; ks < 24; ++ks) {
    v16h a = load_frag_a(H, MLP, row0, ks * 32, lane);
    #pragma unroll
    for (int j = 0; j < 4; ++j) {
      v16h b = load_frag_b(W2, MLP, col0 + j * 16, ks * 32, lane);
      acc[j] = wmma_f16(a, b, acc[j]);
    }
  }
  int rbase = row0 + ((lane & 16) ? 8 : 0);
  #pragma unroll
  for (int j = 0; j < 4; ++j) {
    int f = col0 + j * 16 + (lane & 15);
    float bv = b2[f];
    #pragma unroll
    for (int r = 0; r < 8; ++r) {
      size_t idx = (size_t)(rbase + r) * DIM + f;
      out[idx] = out[idx] + acc[j][r] + bv;
    }
  }
}

extern "C" void kernel_launch(void* const* d_in, const int* in_sizes, int n_in,
                              void* d_out, int out_size, void* d_ws, size_t ws_size,
                              hipStream_t stream) {
  (void)in_sizes; (void)n_in; (void)out_size; (void)ws_size;
  const float* x      = (const float*)d_in[0];
  const float* qkv_w  = (const float*)d_in[1];
  const float* qkv_b  = (const float*)d_in[2];
  const float* proj_w = (const float*)d_in[3];
  const float* proj_b = (const float*)d_in[4];
  const float* btab   = (const float*)d_in[5];
  const float* n1g    = (const float*)d_in[6];
  const float* n1b    = (const float*)d_in[7];
  const float* n2g    = (const float*)d_in[8];
  const float* n2b    = (const float*)d_in[9];
  const float* fc1w   = (const float*)d_in[10];
  const float* fc1b   = (const float*)d_in[11];
  const float* fc2w   = (const float*)d_in[12];
  const float* fc2b   = (const float*)d_in[13];
  float* out = (float*)d_out;

  char* ws = (char*)d_ws;
  size_t off = 0;
  auto alloc = [&](size_t bytes) -> char* {
    char* p = ws + off;
    off += (bytes + 255) & ~(size_t)255;
    return p;
  };
  _Float16* wq16 = (_Float16*)alloc((size_t)576 * 192 * 2);
  _Float16* wp16 = (_Float16*)alloc((size_t)192 * 192 * 2);
  _Float16* wf1  = (_Float16*)alloc((size_t)768 * 192 * 2);
  _Float16* wf2  = (_Float16*)alloc((size_t)192 * 768 * 2);
  _Float16* w16  = (_Float16*)alloc((size_t)NTOK * DIM * 2);              // LN1 windows; reused as attn O
  _Float16* bq   = (_Float16*)alloc((size_t)NWIN * HEADS * WT * HD * 2);  // Q; reused as LN2 out
  _Float16* bk   = (_Float16*)alloc((size_t)NWIN * HEADS * WT * HD * 2);  // K
  _Float16* bv   = (_Float16*)alloc((size_t)NWIN * HEADS * HD * VPAD * 2);// V^T (K-padded)
  _Float16* h16  = (_Float16*)alloc((size_t)NTOK * MLP * 2);              // FC1 out

  // weight conversion fp32 -> f16
  f32_to_f16<<<(576 * 192 + 255) / 256, 256, 0, stream>>>(qkv_w, wq16, 576 * 192);
  f32_to_f16<<<(192 * 192 + 255) / 256, 256, 0, stream>>>(proj_w, wp16, 192 * 192);
  f32_to_f16<<<(768 * 192 + 255) / 256, 256, 0, stream>>>(fc1w, wf1, 768 * 192);
  f32_to_f16<<<(192 * 768 + 255) / 256, 256, 0, stream>>>(fc2w, wf2, 192 * 768);
  hipMemsetAsync(bv, 0, (size_t)NWIN * HEADS * HD * VPAD * 2, stream);

  ln1_shift_kernel<<<NTOK / 4, 128, 0, stream>>>(x, n1g, n1b, w16);
  qkv_gemm<<<dim3(NTOK / 128, 9), 256, 0, stream>>>(w16, wq16, qkv_b, bq, bk, bv);
  attn_kernel<<<NWIN * HEADS, 288, SMEM_ATTN, stream>>>(bq, bk, bv, btab, w16);
  proj_gemm<<<dim3(NTOK / 128, 3), 256, 0, stream>>>(w16, wp16, proj_b, x, out);
  ln2_kernel<<<NTOK / 4, 128, 0, stream>>>(out, n2g, n2b, bq);
  fc1_gemm<<<dim3(NTOK / 128, 12), 256, 0, stream>>>(bq, wf1, fc1b, h16);
  fc2_gemm<<<dim3(NTOK / 128, 3), 256, 0, stream>>>(h16, wf2, fc2b, out);
}
